// Head_30494267801761
// MI455X (gfx1250) — matrix operand order repaired and checked by reading, not compiled, not run on hardware
//
#include <hip/hip_runtime.h>
#include <hip/hip_bf16.h>

// MI455X / gfx1250, wave32. Fused causal attention head with bf16 WMMA.

typedef __attribute__((ext_vector_type(16))) __bf16 v16bf;
typedef __attribute__((ext_vector_type(8)))  __bf16 bf16x8;
typedef __attribute__((ext_vector_type(8)))  float  v8f;

#define B_ 8
#define T_ 2048
#define C_ 1024
#define H_ 64

static __device__ __forceinline__ v8f wmma_bf16(v16bf a, v16bf b, v8f c) {
  // D(f32 16x16) = A(bf16 16x32) * B(bf16 32x16) + C
  return __builtin_amdgcn_wmma_f32_16x16x32_bf16(false, a, false, b, (short)0, c,
                                                 false, false);
}

static __device__ __forceinline__ v16bf join8(bf16x8 lo, bf16x8 hi) {
  return __builtin_shufflevector(lo, hi, 0, 1, 2, 3, 4, 5, 6, 7,
                                 8, 9, 10, 11, 12, 13, 14, 15);
}

// ---------------------------------------------------------------------------
// Kernel 1: convert W[c][h] f32 -> Wt[w][h][c] bf16 (transposed, WMMA-B friendly)
// ---------------------------------------------------------------------------
__global__ void __launch_bounds__(256)
wt_kernel(const float* __restrict__ Wq, const float* __restrict__ Wk,
          const float* __restrict__ Wv, __bf16* __restrict__ Wt) {
  int idx = blockIdx.x * 256 + threadIdx.x;
  if (idx >= 3 * C_ * H_) return;
  int w = idx / (C_ * H_);
  int rem = idx - w * (C_ * H_);
  int c = rem / H_;
  int h = rem - c * H_;
  const float* W = (w == 0) ? Wq : ((w == 1) ? Wk : Wv);
  Wt[(size_t)w * H_ * C_ + (size_t)h * C_ + c] = (__bf16)W[(size_t)c * H_ + h];
}

// ---------------------------------------------------------------------------
// Kernel 2: projections, all three weights in one pass over x (x read ONCE).
// Q,K row-major [B*T][64] bf16; V transposed [B][64][T] bf16.
// One wave = 16 rows of M, N=64 (4 tiles) x 3 weights = 12 f32 accumulators.
// 4-wave blocks x 256 blocks so all WGPs get work (Wt is L0-hot; wave-level
// interleave hides the per-fragment load latency the scheduler won't).
// ---------------------------------------------------------------------------
__global__ void __launch_bounds__(128)
proj_kernel(const float* __restrict__ x, const __bf16* __restrict__ Wt,
            __bf16* __restrict__ Qb, __bf16* __restrict__ Kb,
            __bf16* __restrict__ Vtb) {
  const int lane = threadIdx.x & 31;
  const int wave = threadIdx.x >> 5;
  const int ln = lane & 15;   // A: row within tile; B/D: column
  const int lh = lane >> 4;   // lane half selects K sub-range
  const int m0 = blockIdx.x * 64 + wave * 16;
  const float* xr = x + (size_t)(m0 + ln) * C_;

  v8f acc[12];
#pragma unroll
  for (int i = 0; i < 12; ++i) acc[i] = (v8f){0, 0, 0, 0, 0, 0, 0, 0};

  for (int ki = 0; ki < C_ / 32; ++ki) {
    const int kb = ki * 32 + lh * 16;  // A slots0-15 = K kb..kb+15 (contiguous)
    float4 f0 = *(const float4*)(xr + kb);
    float4 f1 = *(const float4*)(xr + kb + 4);
    float4 f2 = *(const float4*)(xr + kb + 8);
    float4 f3 = *(const float4*)(xr + kb + 12);

    // Preload all 12 B-fragments (distinct registers -> clustered loads).
    const int kb16 = ki * 32 + lh * 16;  // B-frag: 16 contiguous K per lane
    v16bf bfr[12];
#pragma unroll
    for (int w = 0; w < 3; ++w) {
      const __bf16* wp = Wt + (size_t)w * H_ * C_ + (size_t)ln * C_ + kb16;
#pragma unroll
      for (int nt = 0; nt < 4; ++nt) {
        bfr[w * 4 + nt] = *(const v16bf*)(wp + (size_t)nt * 16 * C_);
      }
    }

    v16bf a;
    a[0] = (__bf16)f0.x;  a[1] = (__bf16)f0.y;  a[2] = (__bf16)f0.z;  a[3] = (__bf16)f0.w;
    a[4] = (__bf16)f1.x;  a[5] = (__bf16)f1.y;  a[6] = (__bf16)f1.z;  a[7] = (__bf16)f1.w;
    a[8] = (__bf16)f2.x;  a[9] = (__bf16)f2.y;  a[10] = (__bf16)f2.z; a[11] = (__bf16)f2.w;
    a[12] = (__bf16)f3.x; a[13] = (__bf16)f3.y; a[14] = (__bf16)f3.z; a[15] = (__bf16)f3.w;

#pragma unroll
    for (int i = 0; i < 12; ++i) acc[i] = wmma_bf16(a, bfr[i], acc[i]);
  }

  // D layout: lane -> n (=h within tile), reg r + 8*lh -> m (row).
#pragma unroll
  for (int nt = 0; nt < 4; ++nt) {
    const int h = nt * 16 + ln;
#pragma unroll
    for (int r = 0; r < 8; ++r) {
      const int mrow = m0 + r + 8 * lh;
      Qb[(size_t)mrow * H_ + h] = (__bf16)acc[0 + nt][r];
      Kb[(size_t)mrow * H_ + h] = (__bf16)acc[4 + nt][r];
      const int bb = mrow >> 11;        // / T_
      const int tt = mrow & (T_ - 1);
      Vtb[((size_t)bb * H_ + h) * T_ + tt] = (__bf16)acc[8 + nt][r];
    }
  }
}

// ---------------------------------------------------------------------------
// Kernel 3: flash attention. One wave = 16 queries. Scores computed transposed
// (S^T = K * Q^T) so the f32 D-fragments map 1:1 onto the bf16 A-fragment
// layout needed for P*V (only f32->bf16 converts, no LDS transpose).
// 4-wave blocks x 256 blocks; causal balance: each block holds two `pair`
// tiles and two mirrored `127-pair` tiles -> identical work per block.
// ---------------------------------------------------------------------------
__global__ void __launch_bounds__(128)
attn_kernel(const __bf16* __restrict__ Qb, const __bf16* __restrict__ Kb,
            const __bf16* __restrict__ Vtb, const int* __restrict__ pad,
            float* __restrict__ out) {
  const int lane = threadIdx.x & 31;
  const int wave = threadIdx.x >> 5;              // 0..3
  const int ln = lane & 15;
  const int lh = lane >> 4;
  const int half = blockIdx.x & 1;
  const int pair = blockIdx.x >> 1;               // 0..127
  const int b = wave + 4 * half;                  // 0..7
  const int qt = (wave & 1) ? (127 - pair) : pair;
  const int q0 = qt * 16;
  const int q = q0 + ln;                      // query owned by this lane (softmax stats)

  // Q as WMMA-B operand (Q^T): lane -> query column, h = lh*8..+7 and 16+lh*8..+7 per lane.
  const __bf16* Qrow = Qb + ((size_t)b * T_ + q0 + ln) * H_;
  const v16bf qf0 = join8(*(const bf16x8*)(Qrow + 0 * 32 + lh * 8), *(const bf16x8*)(Qrow + 0 * 32 + 16 + lh * 8));
  const v16bf qf1 = join8(*(const bf16x8*)(Qrow + 1 * 32 + lh * 8), *(const bf16x8*)(Qrow + 1 * 32 + 16 + lh * 8));

  const __bf16* Kbase = Kb + (size_t)b * T_ * H_;
  const __bf16* Vbase = Vtb + (size_t)b * H_ * T_;
  const int* pbase = pad + b * T_;

  float m = -__builtin_inff();
  float l = 0.f;
  v8f o0 = {0, 0, 0, 0, 0, 0, 0, 0};
  v8f o1 = {0, 0, 0, 0, 0, 0, 0, 0};
  v8f o2 = {0, 0, 0, 0, 0, 0, 0, 0};
  v8f o3 = {0, 0, 0, 0, 0, 0, 0, 0};
  const float scale = 0.03125f;               // 1/sqrt(1024)

  const int nChunk = (q0 + 47) >> 5;          // cover keys [0, q0+16) in 32-key chunks
  for (int ch = 0; ch < nChunk; ++ch) {
    const int sbase = ch * 32;

    // Prefetch next chunk's K rows (128B each) and V rows into cache while we
    // compute this chunk (lowers to global_prefetch_b8; L2-resident data).
    if (ch + 1 < nChunk) {
      __builtin_prefetch(Kbase + (size_t)(sbase + 32 + lane) * H_, 0, 3);
      const __bf16* vpre = Vbase + (size_t)lane * T_ + sbase + 32;
      __builtin_prefetch(vpre, 0, 3);
      __builtin_prefetch(vpre + (size_t)32 * T_, 0, 3);
    }

    // --- S^T = K_block * Q^T : two 16x16 D tiles (keys 0-15, 16-31) ---
    v8f st0 = {0, 0, 0, 0, 0, 0, 0, 0};
    v8f st1 = {0, 0, 0, 0, 0, 0, 0, 0};
    {
      const __bf16* k0p = Kbase + (size_t)(sbase + ln) * H_ + lh * 8;
      const __bf16* k1p = Kbase + (size_t)(sbase + 16 + ln) * H_ + lh * 8;
      v16bf a00 = join8(*(const bf16x8*)(k0p), *(const bf16x8*)(k0p + 16));
      v16bf a10 = join8(*(const bf16x8*)(k1p), *(const bf16x8*)(k1p + 16));
      v16bf a01 = join8(*(const bf16x8*)(k0p + 32), *(const bf16x8*)(k0p + 48));
      v16bf a11 = join8(*(const bf16x8*)(k1p + 32), *(const bf16x8*)(k1p + 48));
      st0 = wmma_bf16(a00, qf0, st0);
      st1 = wmma_bf16(a10, qf0, st1);
      st0 = wmma_bf16(a01, qf1, st0);
      st1 = wmma_bf16(a11, qf1, st1);
    }

    float s0[8], s1[8];
#pragma unroll
    for (int r = 0; r < 8; ++r) {
      s0[r] = st0[r] * scale;
      s1[r] = st1[r] * scale;
    }

    // Fast path: chunk fully below diagonal and no padding -> skip masking.
    const int pm = pbase[sbase + lane];
    const bool fastpath = __all(pm != 0) && (sbase + 31 <= q0);
    if (!fastpath) {
#pragma unroll
      for (int r = 0; r < 8; ++r) {
        const int k0 = sbase + r + 8 * lh;    // key of reg r, tile 0
        const int k1 = k0 + 16;               // key of reg r, tile 1
        const bool ok0 = (k0 <= q) && (pbase[k0] != 0);
        const bool ok1 = (k1 <= q) && (pbase[k1] != 0);
        s0[r] = ok0 ? s0[r] : -__builtin_inff();
        s1[r] = ok1 ? s1[r] : -__builtin_inff();
      }
    }

    // --- online softmax (row stats split across lane pairs (ln, ln+16)) ---
    float cm = -__builtin_inff();
#pragma unroll
    for (int r = 0; r < 8; ++r) cm = fmaxf(cm, fmaxf(s0[r], s1[r]));
    cm = fmaxf(cm, __shfl_xor(cm, 16, 32));
    const float mn = fmaxf(m, cm);

    float p0[8], p1[8];
    float rs = 0.f;
#pragma unroll
    for (int r = 0; r < 8; ++r) {
      p0[r] = __expf(s0[r] - mn);
      p1[r] = __expf(s1[r] - mn);
      rs += p0[r] + p1[r];
    }
    rs += __shfl_xor(rs, 16, 32);
    const float alpha = __expf(m - mn);
    l = l * alpha + rs;
    m = mn;

    // P as WMMA-A operand: D-fragment of S^T already matches A layout.
    v16bf pf;
#pragma unroll
    for (int r = 0; r < 8; ++r) {
      pf[r] = (__bf16)p0[r];
      pf[8 + r] = (__bf16)p1[r];
    }

    // Rescale O accumulators: factor per query (O regs are keyed per-query).
#pragma unroll
    for (int r = 0; r < 8; ++r) {
      const float fr = __int_as_float(__builtin_amdgcn_ds_bpermute(
          (r + 8 * lh) << 2, __float_as_int(alpha)));
      o0[r] *= fr; o1[r] *= fr; o2[r] *= fr; o3[r] *= fr;
    }

    // --- O += P * V  (V transposed -> B-fragments from two 16B halves: keys lh*8.., 16+lh*8..) ---
    const __bf16* vp = Vbase + (size_t)ln * T_ + sbase + lh * 8;
    o0 = wmma_bf16(pf, join8(*(const bf16x8*)(vp + (size_t)0 * 16 * T_), *(const bf16x8*)(vp + (size_t)0 * 16 * T_ + 16)), o0);
    o1 = wmma_bf16(pf, join8(*(const bf16x8*)(vp + (size_t)1 * 16 * T_), *(const bf16x8*)(vp + (size_t)1 * 16 * T_ + 16)), o1);
    o2 = wmma_bf16(pf, join8(*(const bf16x8*)(vp + (size_t)2 * 16 * T_), *(const bf16x8*)(vp + (size_t)2 * 16 * T_ + 16)), o2);
    o3 = wmma_bf16(pf, join8(*(const bf16x8*)(vp + (size_t)3 * 16 * T_), *(const bf16x8*)(vp + (size_t)3 * 16 * T_ + 16)), o3);
  }

  // Epilogue: divide by row sum, store f32 output [B][T][64].
  const float linv = 1.f / l;
  float* orow = out + ((size_t)b * T_ + q0) * H_;
#pragma unroll
  for (int r = 0; r < 8; ++r) {
    const float fr = __int_as_float(__builtin_amdgcn_ds_bpermute(
        (r + 8 * lh) << 2, __float_as_int(linv)));
    const int trow = r + 8 * lh;
    orow[(size_t)trow * H_ + 0 * 16 + ln] = o0[r] * fr;
    orow[(size_t)trow * H_ + 1 * 16 + ln] = o1[r] * fr;
    orow[(size_t)trow * H_ + 2 * 16 + ln] = o2[r] * fr;
    orow[(size_t)trow * H_ + 3 * 16 + ln] = o3[r] * fr;
  }
}

// ---------------------------------------------------------------------------
extern "C" void kernel_launch(void* const* d_in, const int* in_sizes, int n_in,
                              void* d_out, int out_size, void* d_ws, size_t ws_size,
                              hipStream_t stream) {
  (void)in_sizes; (void)n_in; (void)out_size; (void)ws_size;
  const float* x  = (const float*)d_in[0];
  const int*   pm = (const int*)d_in[1];
  const float* Wq = (const float*)d_in[2];
  const float* Wk = (const float*)d_in[3];
  const float* Wv = (const float*)d_in[4];
  float* out = (float*)d_out;

  char* wsb = (char*)d_ws;
  __bf16* Wt  = (__bf16*)wsb;                                   // 3*64*1024 bf16
  __bf16* Qb  = (__bf16*)(wsb + (size_t)3 * C_ * H_ * 2);       // [B*T][64]
  __bf16* Kb  = Qb + (size_t)B_ * T_ * H_;                      // [B*T][64]
  __bf16* Vtb = Kb + (size_t)B_ * T_ * H_;                      // [B][64][T]

  wt_kernel<<<(3 * C_ * H_ + 255) / 256, 256, 0, stream>>>(Wq, Wk, Wv, Wt);
  proj_kernel<<<(B_ * T_) / 64, 128, 0, stream>>>(x, Wt, Qb, Kb, Vtb);
  attn_kernel<<<256, 128, 0, stream>>>(Qb, Kb, Vtb, pm, out);
}
